// NetworkGraphEmbeddingConstructor_4604204941423
// MI455X (gfx1250) — matrix operand order repaired and checked by reading, not compiled, not run on hardware
//
#include <hip/hip_runtime.h>

// Problem constants (match reference)
#define NNODES 20000
#define NEDGES 160000
#define FIN    128
#define EMBD   64
#define NHEAD  4
#define CDIM   64
#define HCDIM  256   // NHEAD * CDIM
#define NBLOCKS 5
#define BNEPS  1e-5f

typedef __attribute__((ext_vector_type(16))) __bf16 v16bf;
typedef __attribute__((ext_vector_type(8)))  __bf16 v8bf;
typedef __attribute__((ext_vector_type(8)))  float  v8f;

// ---------- helpers ----------
__device__ __forceinline__ unsigned short f2bf(float f) {
  unsigned int u = __float_as_uint(f);
  u += 0x7FFFu + ((u >> 16) & 1u);          // round to nearest even
  return (unsigned short)(u >> 16);
}

__device__ __forceinline__ void atomicMaxF(float* addr, float val) {
  // classic signed/unsigned trick; addr pre-initialized to -inf
  if (val >= 0.0f) atomicMax((int*)addr, __float_as_int(val));
  else             atomicMin((unsigned int*)addr, __float_as_uint(val));
}

// ---------- elementwise utility kernels ----------
__global__ void k_fill(float* p, float v, int n) {
  int i = blockIdx.x * blockDim.x + threadIdx.x;
  if (i < n) p[i] = v;
}

__global__ void k_f32_to_bf16(const float* __restrict__ s, unsigned short* __restrict__ d, int n) {
  int i = blockIdx.x * blockDim.x + threadIdx.x;
  if (i < n) d[i] = f2bf(s[i]);
}

// Pack weight W[K x Nc] (f32) into per-lane WMMA B-fragment layout (bf16):
// dst[((chunk*tilesN + tn)*32 + lane)*16 + j] = bf16(W[(chunk*32 + (j>>3)*16 + (lane>>4)*8 + (j&7))*Nc + tn*16 + (lane&15)])
__global__ void k_pack_w_frag(const float* __restrict__ W, unsigned short* __restrict__ dst,
                              int K, int Nc) {
  int p = blockIdx.x * blockDim.x + threadIdx.x;
  if (p >= K * Nc) return;
  int j    = p & 15;
  int lane = (p >> 4) & 31;
  int rest = p >> 9;
  int tilesN = Nc >> 4;
  int tn = rest % tilesN;
  int c  = rest / tilesN;
  int k   = c * 32 + (j >> 3) * 16 + (lane >> 4) * 8 + (j & 7);
  int col = tn * 16 + (lane & 15);
  dst[p] = f2bf(W[(size_t)k * Nc + col]);
}

// ---------- WMMA GEMM: Y[M x Nc] = bf16(A[M x K]) @ bf16(Bpack) + bias ----------
// One wave32 per 32x64 output strip (2 M-tiles x 4 N-tiles): each B fragment feeds
// two WMMAs, each A fragment feeds four. 8 f32 accumulators (64 VGPRs).
__global__ void __launch_bounds__(128)
k_wmma_gemm_bias(const unsigned short* __restrict__ A,
                 const unsigned short* __restrict__ Bpack,
                 const float* __restrict__ bias,
                 float* __restrict__ Y,
                 int M, int K, int Nc) {
  int wave = (int)((blockIdx.x * blockDim.x + threadIdx.x) >> 5);
  int lane = threadIdx.x & 31;
  int tilesN = Nc >> 4;
  int groupsN = tilesN >> 2;                 // strips of 4 tiles (64 cols)
  int strips = (M >> 5) * groupsN;           // 32 rows per strip
  if (wave >= strips) return;                // uniform per-wave: EXEC all-1s inside
  int tm = wave / groupsN;
  int tg = wave - tm * groupsN;
  int tn0 = tg * 4;
  int hl  = lane >> 4;                       // lane-group half (0/1)
  int l16 = lane & 15;

  const unsigned short* Arow0 = A + (size_t)(tm * 32 + l16) * K;
  const unsigned short* Arow1 = Arow0 + (size_t)16 * K;

  v8f acc00 = {}, acc01 = {}, acc02 = {}, acc03 = {};
  v8f acc10 = {}, acc11 = {}, acc12 = {}, acc13 = {};
  int nchunks = K >> 5;
  for (int c = 0; c < nchunks; ++c) {
    int kk = c << 5;
    // A 16x32 bf16 fragment: lanes 0-15 K=0..7,16..23 ; lanes 16-31 K=8..15,24..31
    v8bf lo0 = *(const v8bf*)(const void*)(Arow0 + kk + hl * 8);
    v8bf hi0 = *(const v8bf*)(const void*)(Arow0 + kk + 16 + hl * 8);
    v8bf lo1 = *(const v8bf*)(const void*)(Arow1 + kk + hl * 8);
    v8bf hi1 = *(const v8bf*)(const void*)(Arow1 + kk + 16 + hl * 8);
    v16bf a0, a1;
#pragma unroll
    for (int j = 0; j < 8; ++j) {
      a0[j] = lo0[j]; a0[j + 8] = hi0[j];
      a1[j] = lo1[j]; a1[j + 8] = hi1[j];
    }

    const unsigned short* fb =
        Bpack + (((size_t)c * tilesN + tn0) * 32 + lane) * 16;
    if (c + 1 < nchunks) {
      // next chunk's B fragments -> global_prefetch_b8
      __builtin_prefetch(fb + (size_t)tilesN * 32 * 16, 0, 3);
    }
    v16bf b0 = *(const v16bf*)(const void*)(fb);
    v16bf b1 = *(const v16bf*)(const void*)(fb + 32 * 16);
    v16bf b2 = *(const v16bf*)(const void*)(fb + 2 * 32 * 16);
    v16bf b3 = *(const v16bf*)(const void*)(fb + 3 * 32 * 16);

    acc00 = __builtin_amdgcn_wmma_f32_16x16x32_bf16(false, a0, false, b0, (short)0, acc00, false, false);
    acc10 = __builtin_amdgcn_wmma_f32_16x16x32_bf16(false, a1, false, b0, (short)0, acc10, false, false);
    acc01 = __builtin_amdgcn_wmma_f32_16x16x32_bf16(false, a0, false, b1, (short)0, acc01, false, false);
    acc11 = __builtin_amdgcn_wmma_f32_16x16x32_bf16(false, a1, false, b1, (short)0, acc11, false, false);
    acc02 = __builtin_amdgcn_wmma_f32_16x16x32_bf16(false, a0, false, b2, (short)0, acc02, false, false);
    acc12 = __builtin_amdgcn_wmma_f32_16x16x32_bf16(false, a1, false, b2, (short)0, acc12, false, false);
    acc03 = __builtin_amdgcn_wmma_f32_16x16x32_bf16(false, a0, false, b3, (short)0, acc03, false, false);
    acc13 = __builtin_amdgcn_wmma_f32_16x16x32_bf16(false, a1, false, b3, (short)0, acc13, false, false);
  }

#pragma unroll
  for (int t = 0; t < 4; ++t) {
    v8f a0 = (t == 0) ? acc00 : (t == 1) ? acc01 : (t == 2) ? acc02 : acc03;
    v8f a1 = (t == 0) ? acc10 : (t == 1) ? acc11 : (t == 2) ? acc12 : acc13;
    int colB = (tn0 + t) * 16 + l16;
    float bv = bias[colB];
#pragma unroll
    for (int r = 0; r < 8; ++r) {
      // C/D layout: vgpr r -> row = 8*half + r ; col = lane%16
      int row = tm * 32 + hl * 8 + r;
      Y[(size_t)row * Nc + colB]        = a0[r] + bv;
      Y[(size_t)(row + 16) * Nc + colB] = a1[r] + bv;
    }
  }
}

// ---------- attention kernels ----------
// alpha[e,h] = dot(q[dst], k[src] + e_proj) / sqrt(C); running atomic max per (dst,h)
__global__ void k_edge_alpha_max(const int* __restrict__ src, const int* __restrict__ dst,
                                 const float* __restrict__ ea,
                                 const float* __restrict__ q, const float* __restrict__ k,
                                 const float* __restrict__ We, const float* __restrict__ be,
                                 float* __restrict__ alpha, float* __restrict__ amax) {
  int idx = blockIdx.x * blockDim.x + threadIdx.x;
  if (idx >= NEDGES * NHEAD) return;
  int e = idx >> 2;            // / NHEAD
  int h = idx & 3;
  int s = src[e], d = dst[e];
  float w = ea[e];
  const float4* qp  = (const float4*)(q  + (size_t)d * HCDIM + h * CDIM);
  const float4* kp  = (const float4*)(k  + (size_t)s * HCDIM + h * CDIM);
  const float4* Wep = (const float4*)(We + h * CDIM);
  const float4* bep = (const float4*)(be + h * CDIM);
  float acc = 0.0f;
#pragma unroll 4
  for (int c = 0; c < CDIM / 4; ++c) {
    float4 qv = qp[c], kv = kp[c], wv = Wep[c], bv = bep[c];
    acc += qv.x * (kv.x + w * wv.x + bv.x);
    acc += qv.y * (kv.y + w * wv.y + bv.y);
    acc += qv.z * (kv.z + w * wv.z + bv.z);
    acc += qv.w * (kv.w + w * wv.w + bv.w);
  }
  acc *= 0.125f;               // 1/sqrt(64)
  alpha[idx] = acc;
  atomicMaxF(amax + (size_t)d * NHEAD + h, acc);
}

__global__ void k_edge_exp_den(const int* __restrict__ dst,
                               const float* __restrict__ alpha,
                               const float* __restrict__ amax,
                               float* __restrict__ exl, float* __restrict__ den) {
  int idx = blockIdx.x * blockDim.x + threadIdx.x;
  if (idx >= NEDGES * NHEAD) return;
  int e = idx >> 2;
  int h = idx & 3;
  int d = dst[e];
  float ex = __expf(alpha[idx] - amax[(size_t)d * NHEAD + h]);
  exl[idx] = ex;
  atomicAdd(den + (size_t)d * NHEAD + h, ex);
}

// out[dst, hc] += (v[src, hc] + e_proj[hc]) * attn[e, h]   (thread per edge*channel)
__global__ void k_edge_scatter(const int* __restrict__ src, const int* __restrict__ dst,
                               const float* __restrict__ ea,
                               const float* __restrict__ v,
                               const float* __restrict__ We, const float* __restrict__ be,
                               const float* __restrict__ exl, const float* __restrict__ den,
                               float* __restrict__ out) {
  long long idx = (long long)blockIdx.x * blockDim.x + threadIdx.x;
  if (idx >= (long long)NEDGES * HCDIM) return;
  int e  = (int)(idx >> 8);    // / HCDIM
  int hc = (int)(idx & 255);
  int h  = hc >> 6;            // / CDIM
  int s = src[e], d = dst[e];
  float attn = exl[(size_t)e * NHEAD + h] /
               fmaxf(den[(size_t)d * NHEAD + h], 1e-16f);
  float vj = v[(size_t)s * HCDIM + hc] + ea[e] * We[hc] + be[hc];
  atomicAdd(out + (size_t)d * HCDIM + hc, vj * attn);
}

// ---------- beta gate ----------
__global__ void k_beta_logit(const float* __restrict__ out, const float* __restrict__ skip,
                             const float* __restrict__ Wb, float* __restrict__ beta) {
  int n = blockIdx.x * blockDim.x + threadIdx.x;
  if (n >= NNODES) return;
  const float4* op = (const float4*)(out  + (size_t)n * HCDIM);
  const float4* sp = (const float4*)(skip + (size_t)n * HCDIM);
  const float4* w0 = (const float4*)(Wb);
  const float4* w1 = (const float4*)(Wb + HCDIM);
  const float4* w2 = (const float4*)(Wb + 2 * HCDIM);
  float acc = 0.0f;
#pragma unroll 4
  for (int i = 0; i < HCDIM / 4; ++i) {
    float4 o = op[i], s = sp[i], a = w0[i], b = w1[i], c = w2[i];
    acc += o.x * a.x + s.x * b.x + (o.x - s.x) * c.x;
    acc += o.y * a.y + s.y * b.y + (o.y - s.y) * c.y;
    acc += o.z * a.z + s.z * b.z + (o.z - s.z) * c.z;
    acc += o.w * a.w + s.w * b.w + (o.w - s.w) * c.w;
  }
  beta[n] = 1.0f / (1.0f + __expf(-acc));
}

__global__ void k_combine_bf16(const float* __restrict__ out, const float* __restrict__ skip,
                               const float* __restrict__ beta, unsigned short* __restrict__ comb) {
  int idx = blockIdx.x * blockDim.x + threadIdx.x;
  if (idx >= NNODES * HCDIM) return;
  int n = idx >> 8;
  float bt = beta[n];
  comb[idx] = f2bf(bt * skip[idx] + (1.0f - bt) * out[idx]);
}

// ---------- relu + batch norm ----------
__global__ void k_relu(float* __restrict__ x, int n) {
  int i = blockIdx.x * blockDim.x + threadIdx.x;
  if (i < n) x[i] = fmaxf(x[i], 0.0f);
}

__global__ void k_colstats(const float* __restrict__ X, float* __restrict__ mean,
                           float* __restrict__ var) {
  __shared__ float ssum[256];
  __shared__ float ssq[256];
  int c = blockIdx.x;                    // one block per column (EMBD blocks)
  float s = 0.0f, s2 = 0.0f;
  for (int r = threadIdx.x; r < NNODES; r += blockDim.x) {
    float x = X[(size_t)r * EMBD + c];
    s += x; s2 += x * x;
  }
  ssum[threadIdx.x] = s; ssq[threadIdx.x] = s2;
  __syncthreads();
  for (int off = 128; off > 0; off >>= 1) {
    if (threadIdx.x < off) {
      ssum[threadIdx.x] += ssum[threadIdx.x + off];
      ssq[threadIdx.x]  += ssq[threadIdx.x + off];
    }
    __syncthreads();
  }
  if (threadIdx.x == 0) {
    float m = ssum[0] * (1.0f / NNODES);
    mean[c] = m;
    var[c]  = ssq[0] * (1.0f / NNODES) - m * m;   // biased, matches torch BN
  }
}

__global__ void k_bn_apply(const float* __restrict__ X, const float* __restrict__ mean,
                           const float* __restrict__ var, const float* __restrict__ g,
                           const float* __restrict__ b, float* __restrict__ outF,
                           unsigned short* __restrict__ outBF) {
  int idx = blockIdx.x * blockDim.x + threadIdx.x;
  if (idx >= NNODES * EMBD) return;
  int c = idx & (EMBD - 1);
  float y = (X[idx] - mean[c]) * rsqrtf(var[c] + BNEPS) * g[c] + b[c];
  outF[idx]  = y;
  outBF[idx] = f2bf(y);                 // bf16 activation for next block's WMMA GEMMs
}

// ---------- host ----------
extern "C" void kernel_launch(void* const* d_in, const int* in_sizes, int n_in,
                              void* d_out, int out_size, void* d_ws, size_t ws_size,
                              hipStream_t stream) {
  (void)in_sizes; (void)n_in; (void)out_size; (void)ws_size;
  const float* x   = (const float*)d_in[0];       // [N, 128]
  const int*   ei  = (const int*)d_in[1];         // [2, E] (src row, dst row)
  const float* ea  = (const float*)d_in[2];       // [E, 1]
  const int* src = ei;
  const int* dst = ei + NEDGES;

  // params flattened in setup_inputs() insertion order:
  // per block: conv{qW,qb,kW,kb,vW,vb,eW,eb,skipW,skipb,betaW}, transfW, transfb, bnG, bnB
  auto P = [&](int blk, int leaf) -> const float* {
    return (const float*)d_in[3 + blk * 15 + leaf];
  };

  // workspace carve-up (~140 MB required)
  char* w = (char*)d_ws;
  auto carve = [&](size_t bytes) -> void* {
    void* p = (void*)w;
    w += (bytes + 255) & ~(size_t)255;
    return p;
  };
  float* qb    = (float*)carve((size_t)NNODES * HCDIM * 4);
  float* kb    = (float*)carve((size_t)NNODES * HCDIM * 4);
  float* vb    = (float*)carve((size_t)NNODES * HCDIM * 4);
  float* skipb = (float*)carve((size_t)NNODES * HCDIM * 4);
  float* aggb  = (float*)carve((size_t)NNODES * HCDIM * 4);
  float* alpha = (float*)carve((size_t)NEDGES * NHEAD * 4);
  float* exl   = (float*)carve((size_t)NEDGES * NHEAD * 4);
  float* amax  = (float*)carve((size_t)NNODES * NHEAD * 4);
  float* den   = (float*)carve((size_t)NNODES * NHEAD * 4);
  float* beta  = (float*)carve((size_t)NNODES * 4);
  float* hpre  = (float*)carve((size_t)NNODES * EMBD * 4);
  float* hnext = (float*)carve((size_t)NNODES * EMBD * 4);
  float* meanb = (float*)carve(EMBD * 4);
  float* varb  = (float*)carve(EMBD * 4);
  unsigned short* act_bf  = (unsigned short*)carve((size_t)NNODES * HCDIM * 2);
  unsigned short* comb_bf = (unsigned short*)carve((size_t)NNODES * HCDIM * 2);
  unsigned short* w_bf    = (unsigned short*)carve((size_t)HCDIM * HCDIM * 2);

  const int TB = 256;
  auto grid1 = [](long long n, int tb) { return dim3((unsigned)((n + tb - 1) / tb)); };

  auto gemm = [&](const unsigned short* Abf, const float* W, const float* bias,
                  float* Y, int M, int K, int Nc) {
    k_pack_w_frag<<<grid1((long long)K * Nc, TB), TB, 0, stream>>>(W, w_bf, K, Nc);
    int strips = (M / 32) * (Nc / 64);
    int blocks = (strips + 3) / 4;              // 4 waves (128 threads) per block
    k_wmma_gemm_bias<<<dim3(blocks), dim3(128), 0, stream>>>(Abf, w_bf, bias, Y, M, K, Nc);
  };

  // initial activation -> bf16
  k_f32_to_bf16<<<grid1((long long)NNODES * FIN, TB), TB, 0, stream>>>(x, act_bf, NNODES * FIN);

  for (int blk = 0; blk < NBLOCKS; ++blk) {
    const int D = (blk == 0) ? FIN : EMBD;
    const float *Wq = P(blk, 0),  *bq = P(blk, 1);
    const float *Wk = P(blk, 2),  *bk = P(blk, 3);
    const float *Wv = P(blk, 4),  *bv = P(blk, 5);
    const float *We = P(blk, 6),  *be = P(blk, 7);
    const float *Ws = P(blk, 8),  *bs = P(blk, 9);
    const float *Wbeta = P(blk, 10);
    const float *Wt = P(blk, 11), *bt = P(blk, 12);
    const float *bg = P(blk, 13), *bb = P(blk, 14);

    // dense projections via WMMA
    gemm(act_bf, Wq, bq, qb,    NNODES, D, HCDIM);
    gemm(act_bf, Wk, bk, kb,    NNODES, D, HCDIM);
    gemm(act_bf, Wv, bv, vb,    NNODES, D, HCDIM);
    gemm(act_bf, Ws, bs, skipb, NNODES, D, HCDIM);

    // attention (segment softmax + scatter)
    k_fill<<<grid1((long long)NNODES * NHEAD, TB), TB, 0, stream>>>(amax, -__builtin_inff(), NNODES * NHEAD);
    k_fill<<<grid1((long long)NNODES * NHEAD, TB), TB, 0, stream>>>(den, 0.0f, NNODES * NHEAD);
    k_fill<<<grid1((long long)NNODES * HCDIM, TB), TB, 0, stream>>>(aggb, 0.0f, NNODES * HCDIM);
    k_edge_alpha_max<<<grid1((long long)NEDGES * NHEAD, TB), TB, 0, stream>>>(
        src, dst, ea, qb, kb, We, be, alpha, amax);
    k_edge_exp_den<<<grid1((long long)NEDGES * NHEAD, TB), TB, 0, stream>>>(
        dst, alpha, amax, exl, den);
    k_edge_scatter<<<grid1((long long)NEDGES * HCDIM, TB), TB, 0, stream>>>(
        src, dst, ea, vb, We, be, exl, den, aggb);

    // beta-gated skip -> bf16 combined activation
    k_beta_logit<<<grid1(NNODES, TB), TB, 0, stream>>>(aggb, skipb, Wbeta, beta);
    k_combine_bf16<<<grid1((long long)NNODES * HCDIM, TB), TB, 0, stream>>>(
        aggb, skipb, beta, comb_bf);

    // transf GEMM + relu + batch norm
    gemm(comb_bf, Wt, bt, hpre, NNODES, HCDIM, EMBD);
    k_relu<<<grid1((long long)NNODES * EMBD, TB), TB, 0, stream>>>(hpre, NNODES * EMBD);
    k_colstats<<<dim3(EMBD), dim3(256), 0, stream>>>(hpre, meanb, varb);
    float* outF = (blk == NBLOCKS - 1) ? (float*)d_out : hnext;
    k_bn_apply<<<grid1((long long)NNODES * EMBD, TB), TB, 0, stream>>>(
        hpre, meanb, varb, bg, bb, outF, act_bf);
  }
}
